// RGCNEncoder_61675730371074
// MI455X (gfx1250) — matrix operand order repaired and checked by reading, not compiled, not run on hardware
//
#include <hip/hip_runtime.h>
#include <hip/hip_bf16.h>

typedef __attribute__((ext_vector_type(16))) __bf16 v16bf;
typedef __attribute__((ext_vector_type(8)))  float  v8f;

constexpr int DIM   = 768;
constexpr int NREL  = 7;
constexpr int NL    = 5;
constexpr int KTOT  = 8 * DIM;       // 6144: [h | mean_0..mean_6]
constexpr int KPHASE = DIM / 32;     // 24 K-steps per 768-wide source block

// ---------------------------------------------------------------------------
// Build B in bf16, K-contiguous ("column-major"): Wt[l][n][k], k<768 -> root,
// else W[r=k/768-1]. One-time cost per launch.
// ---------------------------------------------------------------------------
__global__ void convert_w_kernel(const float* __restrict__ weights,  // [NL][7][D][D]
                                 const float* __restrict__ roots,    // [NL][D][D]
                                 __bf16* __restrict__ Wt,            // [NL][D][KTOT]
                                 size_t total)
{
    size_t idx = (size_t)blockIdx.x * blockDim.x + threadIdx.x;
    if (idx >= total) return;
    int k = (int)(idx % KTOT);
    size_t rem = idx / KTOT;
    int n = (int)(rem % DIM);
    int l = (int)(rem / DIM);
    float v;
    if (k < DIM) {
        v = roots[((size_t)l * DIM + k) * DIM + n];
    } else {
        int r  = k / DIM - 1;
        int kk = k % DIM;
        v = weights[((((size_t)l * NREL + r) * DIM) + kk) * DIM + n];
    }
    Wt[idx] = (__bf16)v;
}

// ---------------------------------------------------------------------------
// Per-relation in-degree counts (float, exact for small ints)
// ---------------------------------------------------------------------------
__global__ void count_kernel(const int* __restrict__ ei, const int* __restrict__ et,
                             float* __restrict__ cnt, int E)
{
    int e = blockIdx.x * blockDim.x + threadIdx.x;
    if (e >= E) return;
    int d = ei[E + e];
    int r = et[e] + 1;               // [-1..5] -> [0..6]
    unsafeAtomicAdd(&cnt[d * NREL + r], 1.0f);
}

// ---------------------------------------------------------------------------
// Scatter h[src] into aggsum[dst][rel][:]  (fp32 atomics, float4 chunks)
// ---------------------------------------------------------------------------
__global__ void scatter_kernel(const float* __restrict__ hin,
                               const int* __restrict__ ei, const int* __restrict__ et,
                               float* __restrict__ aggsum, int E)
{
    constexpr int CH = DIM / 4;      // 192 float4 chunks per edge
    long long tid = (long long)blockIdx.x * blockDim.x + threadIdx.x;
    int e = (int)(tid / CH);
    int c = (int)(tid % CH);
    if (e >= E) return;
    int s = ei[e];
    int d = ei[E + e];
    int r = et[e] + 1;
    const float4 v = *(const float4*)(hin + (size_t)s * DIM + c * 4);
    float* dst = aggsum + ((size_t)d * NREL + r) * DIM + c * 4;
    unsafeAtomicAdd(dst + 0, v.x);
    unsafeAtomicAdd(dst + 1, v.y);
    unsafeAtomicAdd(dst + 2, v.z);
    unsafeAtomicAdd(dst + 3, v.w);
}

__global__ void invcnt_kernel(const float* __restrict__ cnt, float* __restrict__ icnt, int n7)
{
    int i = blockIdx.x * blockDim.x + threadIdx.x;
    if (i >= n7) return;
    icnt[i] = 1.0f / fmaxf(cnt[i], 1.0f);
}

// ---------------------------------------------------------------------------
// Fused GEMM: out[row][col] = relu( bias[col] + sum_k A[row][k] * B[k][col] )
//   A = [h | mean_r] with the 1/cnt scaling fused into the LDS fill.
// Block = 128x128 tile, 8 waves, each wave 32x64 (2x4 WMMA accumulators).
// K loop = 8 phases (source block) x 24 steps: source pointer and scale are
// phase-invariant -> no division / branching / icnt reload in the hot loop.
// Grid: x = column panel (6), y = row strip -> panels of one strip run
// concurrently and share the fp32 A strip through L2.
// ---------------------------------------------------------------------------
__global__ __launch_bounds__(256)
void rgcn_gemm_kernel(const float* __restrict__ hin,      // [N][D]
                      const float* __restrict__ aggsum,   // [N][7][D]
                      const float* __restrict__ icnt,     // [N][7]
                      const __bf16* __restrict__ Bt,      // [D][KTOT] (K-contiguous)
                      const float* __restrict__ bias,     // [D]
                      float* __restrict__ hout, int n)
{
    __shared__ __align__(32) __bf16 As[8 * 32 * 16];   // 8 KB, fragment order
    __shared__ __align__(32) __bf16 Bs[8 * 32 * 16];   // 8 KB, fragment order

    const int tid  = threadIdx.x;
    const int lane = tid & 31;
    const int sub  = tid >> 5;             // fill: subtile 0..7 ; compute: wave id

    const int blockN = blockIdx.x * 128;   // column panel
    const int blockM = blockIdx.y * 128;   // row strip

    // ---- fill-side mapping (thread fills one A fragment + one B fragment) ----
    const int  a_row = blockM + sub * 16 + (lane & 15);
    const bool valid = a_row < n;
    const int  kh    = lane >> 4;                       // K-half of this lane
    const int  b_col = blockN + sub * 16 + (lane & 15); // global output column
    const __bf16* bsrc = Bt + (size_t)b_col * KTOT + 16 * kh;

    __bf16* aDst = &As[(sub * 32 + lane) * 16];
    __bf16* bDst = &Bs[(sub * 32 + lane) * 16];

    // ---- compute-side mapping ----
    const int wm = sub & 3;        // 4 M-groups
    const int wn = sub >> 2;       // 2 N-groups
    const int t0 = wm * 2;         // first M subtile of this wave
    const int u0 = wn * 4;         // first N subtile of this wave

    v8f acc[2][4];
    #pragma unroll
    for (int i = 0; i < 2; ++i)
        #pragma unroll
        for (int j = 0; j < 4; ++j)
            acc[i][j] = v8f{0.f, 0.f, 0.f, 0.f, 0.f, 0.f, 0.f, 0.f};

    for (int phase = 0; phase < 8; ++phase) {
        // phase-invariant A source pointer + mean scale
        const float* ap;
        float scale;
        if (phase == 0) {
            ap = hin + (size_t)a_row * DIM + 16 * kh;
            scale = 1.0f;
        } else {
            const int rr = phase - 1;
            ap = aggsum + ((size_t)a_row * NREL + rr) * DIM + 16 * kh;
            scale = valid ? icnt[a_row * NREL + rr] : 1.0f;
        }
        const __bf16* bp = bsrc + (size_t)phase * DIM;

        for (int kk = 0; kk < KPHASE; ++kk) {
            // ---------------- global loads (before barriers) ----------------
            float4 a0{}, a1{}, a2{}, a3{};
            if (valid) {
                const float* p = ap + kk * 32;
                a0 = *(const float4*)(p + 0);
                a1 = *(const float4*)(p + 4);
                a2 = *(const float4*)(p + 8);
                a3 = *(const float4*)(p + 12);
            }
            const v16bf bvec = *(const v16bf*)(bp + kk * 32);

            __syncthreads();   // previous iteration's compute done

            // ---------------- LDS store in fragment order ----------------
            const float av[16] = {a0.x, a0.y, a0.z, a0.w, a1.x, a1.y, a1.z, a1.w,
                                  a2.x, a2.y, a2.z, a2.w, a3.x, a3.y, a3.z, a3.w};
            #pragma unroll
            for (int e2 = 0; e2 < 16; ++e2)
                aDst[e2] = (__bf16)(av[e2] * scale);
            *(v16bf*)bDst = bvec;

            __syncthreads();

            // ---------------- WMMA compute ----------------
            v16bf af[2], bf[4];
            #pragma unroll
            for (int ti = 0; ti < 2; ++ti)
                af[ti] = *(const v16bf*)&As[((t0 + ti) * 32 + lane) * 16];
            #pragma unroll
            for (int ui = 0; ui < 4; ++ui)
                bf[ui] = *(const v16bf*)&Bs[((u0 + ui) * 32 + lane) * 16];

            #pragma unroll
            for (int ti = 0; ti < 2; ++ti)
                #pragma unroll
                for (int ui = 0; ui < 4; ++ui)
                    acc[ti][ui] = __builtin_amdgcn_wmma_f32_16x16x32_bf16(
                        false, af[ti], false, bf[ui],
                        (short)0, acc[ti][ui], false, false);
        }
    }

    // ---------------- epilogue: bias + ReLU, fp32 store ----------------
    // C layout: lane L -> N = L%16, M = vgpr + 8*(L/16)
    #pragma unroll
    for (int ui = 0; ui < 4; ++ui) {
        const int col = blockN + (u0 + ui) * 16 + (lane & 15);
        const float bv = bias[col];
        #pragma unroll
        for (int ti = 0; ti < 2; ++ti) {
            const int rbase = blockM + (t0 + ti) * 16 + 8 * (lane >> 4);
            #pragma unroll
            for (int v = 0; v < 8; ++v) {
                const int row = rbase + v;
                if (row < n) {
                    const float val = acc[ti][ui][v] + bv;
                    hout[(size_t)row * DIM + col] = val > 0.f ? val : 0.f;
                }
            }
        }
    }
}

// ---------------------------------------------------------------------------
extern "C" void kernel_launch(void* const* d_in, const int* in_sizes, int n_in,
                              void* d_out, int out_size, void* d_ws, size_t ws_size,
                              hipStream_t stream)
{
    const float* x       = (const float*)d_in[0];   // [N][D]
    const int*   ei      = (const int*)  d_in[1];   // [2][E]
    const int*   et      = (const int*)  d_in[2];   // [E]
    const float* weights = (const float*)d_in[3];   // [NL][7][D][D]
    const float* roots   = (const float*)d_in[4];   // [NL][D][D]
    const float* biases  = (const float*)d_in[5];   // [NL][D]
    float* out = (float*)d_out;

    const int N = in_sizes[0] / DIM;   // 20000
    const int E = in_sizes[2];         // 100000

    // workspace carve-out
    char* ws = (char*)d_ws;
    size_t off = 0;
    auto carve = [&](size_t bytes) -> char* {
        char* p = ws + off;
        off = (off + bytes + 255) & ~(size_t)255;
        return p;
    };
    __bf16* Wt     = (__bf16*)carve((size_t)NL * DIM * KTOT * sizeof(__bf16)); // 47 MB
    float*  hA     = (float*) carve((size_t)N * DIM * sizeof(float));          // 61 MB
    float*  hB     = (float*) carve((size_t)N * DIM * sizeof(float));          // 61 MB
    float*  aggsum = (float*) carve((size_t)N * NREL * DIM * sizeof(float));   // 430 MB
    float*  cnt    = (float*) carve((size_t)N * NREL * sizeof(float));
    float*  icnt   = (float*) carve((size_t)N * NREL * sizeof(float));
    (void)ws_size; (void)n_in; (void)out_size;

    // one-time bf16 weight repack (B matrices, K-contiguous)
    {
        size_t total = (size_t)NL * DIM * KTOT;
        convert_w_kernel<<<(unsigned)((total + 255) / 256), 256, 0, stream>>>(
            weights, roots, Wt, total);
    }

    const float* cur = x;
    for (int l = 0; l < NL; ++l) {
        hipMemsetAsync(aggsum, 0, (size_t)N * NREL * DIM * sizeof(float), stream);
        hipMemsetAsync(cnt,    0, (size_t)N * NREL * sizeof(float),       stream);

        count_kernel<<<(E + 255) / 256, 256, 0, stream>>>(ei, et, cnt, E);

        long long sc_threads = (long long)E * (DIM / 4);
        scatter_kernel<<<(unsigned)((sc_threads + 255) / 256), 256, 0, stream>>>(
            cur, ei, et, aggsum, E);

        invcnt_kernel<<<(N * NREL + 255) / 256, 256, 0, stream>>>(cnt, icnt, N * NREL);

        float* dst = (l == NL - 1) ? out : ((l & 1) ? hB : hA);
        dim3 grid(DIM / 128, (N + 127) / 128);   // x = column panel, y = row strip
        rgcn_gemm_kernel<<<grid, 256, 0, stream>>>(
            cur, aggsum, icnt, Wt + (size_t)l * DIM * KTOT, biases + (size_t)l * DIM,
            dst, N);
        cur = dst;
    }
}
